// LocalAttentionBlock_73177652789838
// MI455X (gfx1250) — matrix operand order repaired and checked by reading, not compiled, not run on hardware
//
#include <hip/hip_runtime.h>

// ---------------------------------------------------------------------------
// Transformer block (local/banded attention) for MI455X (gfx1250, wave32).
// GEMMs: TDM (tensor_load_to_lds) double-buffered LDS staging +
//        V_WMMA_F32_16X16X32_BF16 with fp32 accumulate.
// Block tile 128x128, K-step 64; per-wave 32x64 register block.
// Attention / LayerNorm / softmax in fp32.
// ---------------------------------------------------------------------------

typedef __bf16 v16bf __attribute__((ext_vector_type(16)));
typedef float  v8f   __attribute__((ext_vector_type(8)));
typedef unsigned int v4u __attribute__((ext_vector_type(4)));
typedef int          v8i __attribute__((ext_vector_type(8)));
typedef int          v4i __attribute__((ext_vector_type(4)));

union FragBF16 { v16bf v; unsigned int u[8]; };

__device__ __forceinline__ unsigned short f2bf(float f) {
    unsigned int u = __float_as_uint(f);
    u += 0x7FFFu + ((u >> 16) & 1u);   // round-to-nearest-even
    return (unsigned short)(u >> 16);
}

// ---------------------------------------------------------------------------
// fp32 -> bf16 elementwise cast
// ---------------------------------------------------------------------------
__global__ void lab_cast_bf16(const float* __restrict__ in,
                              unsigned short* __restrict__ out, int n) {
    int i = blockIdx.x * blockDim.x + threadIdx.x;
    if (i < n) out[i] = f2bf(in[i]);
}

// ---------------------------------------------------------------------------
// TDM: issue a 2-D tile load (tileRows x tileCols bf16, row stride K elements)
// from global into LDS (rows packed contiguously).  D# per CDNA5 ISA ch.8:
// group0 = {count, lds_addr, global_addr, type}, group1 = {data_size, tensor
// dims, tile dims, strides}; groups 2/3 + extra operand unused (zero).
// ---------------------------------------------------------------------------
__device__ __forceinline__ void tdm_load_tile_2d(const unsigned short* gsrc,
                                                 const unsigned short* ldst,
                                                 int K, int tensorRows,
                                                 int tileRows, int tileCols) {
    unsigned long long ga = (unsigned long long)(uintptr_t)gsrc;
    unsigned int lds = (unsigned int)(uintptr_t)ldst;   // addr[31:0] = LDS offset

    v4u g0;
    g0[0] = 1u;                                          // count=1, user mode
    g0[1] = lds;                                         // lds_addr
    g0[2] = (unsigned int)ga;                            // global_addr[31:0]
    g0[3] = ((unsigned int)(ga >> 32) & 0x01FFFFFFu)     // global_addr[56:32]
            | (2u << 30);                                // type = 2 ("image")

    const unsigned int td0 = (unsigned int)K;            // tensor_dim0
    const unsigned int td1 = (unsigned int)tensorRows;   // tensor_dim1
    v8i g1;
    g1[0] = (int)(1u << 16);                             // data_size = 2 bytes
    g1[1] = (int)((td0 & 0xFFFFu) << 16);                // dim0[15:0] @ [63:48]
    g1[2] = (int)(((td0 >> 16) & 0xFFFFu)                // dim0[31:16]
            | ((td1 & 0xFFFFu) << 16));                  // dim1[15:0]
    g1[3] = (int)(((td1 >> 16) & 0xFFFFu)                // dim1[31:16]
            | (((unsigned int)tileCols & 0xFFFFu) << 16)); // tile_dim0
    g1[4] = (int)((unsigned int)tileRows & 0xFFFFu);     // tile_dim1, tile_dim2=0
    g1[5] = (int)(unsigned int)K;                        // dim0_stride[31:0]
    g1[6] = 0;                                           // stride hi / dim1_stride lo
    g1[7] = 0;

    v4i z4 = {0, 0, 0, 0};
    v8i z8 = {0, 0, 0, 0, 0, 0, 0, 0};
    __builtin_amdgcn_tensor_load_to_lds(g0, g1, z4, z4, z8, /*cpol=*/0);
}

// ---------------------------------------------------------------------------
// TDM + LDS double-buffered WMMA GEMM:
//   out[M,N] = A[M,K](bf16) * W[N,K]^T(bf16) + bias;  epi==1: GELU -> bf16.
// Block tile 128(M) x 128(N), staged K-step 64.  256 threads = 8 waves in a
// 4(M) x 2(N) grid; each wave computes 32x64 (2x4 WMMA tiles): one B-frag
// feeds 2 WMMAs, one A-frag feeds 4.  Wave 0 drives the TDM pipeline.
// ---------------------------------------------------------------------------
__global__ void __launch_bounds__(256)
lab_gemm_wmma_tdm(const unsigned short* __restrict__ A,
                  const unsigned short* __restrict__ Bw,
                  const float* __restrict__ bias,
                  float* __restrict__ outF,
                  unsigned short* __restrict__ outB,
                  int M, int N, int K, int epi) {
    __shared__ unsigned short sA[2][128 * 64];   // 16 KB per buffer
    __shared__ unsigned short sB[2][128 * 64];   // 16 KB per buffer

    const int lane = threadIdx.x & 31;
    const int wave = threadIdx.x >> 5;
    const int half = lane >> 4;       // 0: lanes 0-15, 1: lanes 16-31
    const int l15  = lane & 15;
    const int tm   = (wave & 3) * 32;     // wave's M offset in block tile
    const int tn   = (wave >> 2) * 64;    // wave's N offset in block tile
    const int blockM = blockIdx.x * 128;
    const int blockN = blockIdx.y * 128;
    const int NK = K / 64;

    v8f acc[2][4];
#pragma unroll
    for (int mt = 0; mt < 2; ++mt)
#pragma unroll
        for (int t = 0; t < 4; ++t)
            acc[mt][t] = (v8f){0.f,0.f,0.f,0.f,0.f,0.f,0.f,0.f};

    if (wave == 0) {   // prologue: stage k-step 0 into buffer 0
        tdm_load_tile_2d(A  + (size_t)blockM * K, &sA[0][0], K, M - blockM, 128, 64);
        tdm_load_tile_2d(Bw + (size_t)blockN * K, &sB[0][0], K, N - blockN, 128, 64);
    }

    for (int ks = 0; ks < NK; ++ks) {
        const int p = ks & 1;
        if (wave == 0) {
            if (ks + 1 < NK) {   // prefetch next k-step into the other buffer
                const int k1 = (ks + 1) * 64;
                tdm_load_tile_2d(A  + (size_t)blockM * K + k1, &sA[p ^ 1][0],
                                 K, M - blockM, 128, 64);
                tdm_load_tile_2d(Bw + (size_t)blockN * K + k1, &sB[p ^ 1][0],
                                 K, N - blockN, 128, 64);
                __builtin_amdgcn_s_wait_tensorcnt(2);  // pair for buf p done
            } else {
                __builtin_amdgcn_s_wait_tensorcnt(0);
            }
        }
        __syncthreads();   // buffer p visible to all 8 waves

#pragma unroll
        for (int kk = 0; kk < 64; kk += 32) {
            // A fragments (16x32 each): lane holds row tm + mt*16 + l15;
            //   half==0: u[0..3]=K0..7,  u[4..7]=K16..23
            //   half==1: u[0..3]=K8..15, u[4..7]=K24..31
            FragBF16 af[2];
#pragma unroll
            for (int mt = 0; mt < 2; ++mt) {
                const unsigned short* aRow =
                    &sA[p][(tm + mt * 16 + l15) * 64 + kk];
                *(uint4*)&af[mt].u[0] = *(const uint4*)(aRow + half * 8);
                *(uint4*)&af[mt].u[4] = *(const uint4*)(aRow + 16 + half * 8);
            }
#pragma unroll
            for (int t = 0; t < 4; ++t) {
                // B fragment (32x16): lane holds col tn+t*16+l15, K half*8 / 16+half*8
                const unsigned short* bRow =
                    &sB[p][(tn + t * 16 + l15) * 64 + kk];
                FragBF16 bf;
                *(uint4*)&bf.u[0] = *(const uint4*)(bRow + half * 8);
                *(uint4*)&bf.u[4] = *(const uint4*)(bRow + 16 + half * 8);
#pragma unroll
                for (int mt = 0; mt < 2; ++mt)
                    acc[mt][t] = __builtin_amdgcn_wmma_f32_16x16x32_bf16(
                                     false, af[mt].v, false, bf.v, (short)0,
                                     acc[mt][t], false, false);
            }
        }
        __syncthreads();   // all waves done with buffer p before reuse
    }

    // C/D layout: VGPR r -> M = row0 + r, N = col
#pragma unroll
    for (int mt = 0; mt < 2; ++mt) {
        const int row0 = blockM + tm + mt * 16 + half * 8;
#pragma unroll
        for (int t = 0; t < 4; ++t) {
            const int   col = blockN + tn + t * 16 + l15;
            const float bv  = bias ? bias[col] : 0.f;
#pragma unroll
            for (int r = 0; r < 8; ++r) {
                float v = acc[mt][t][r] + bv;
                if (epi == 1) {
                    v = 0.5f * v * (1.0f + erff(v * 0.70710678118654752f));
                    outB[(size_t)(row0 + r) * N + col] = f2bf(v);
                } else {
                    outF[(size_t)(row0 + r) * N + col] = v;
                }
            }
        }
    }
}

// ---------------------------------------------------------------------------
// Banded (window) attention.  One thread per (head, query); two-pass softmax
// over keys in [i-w, i+w].  qkv is fp32 [L, 3D]; ctx written as bf16 [L, D].
// ---------------------------------------------------------------------------
#define LAB_HD 64

__global__ void lab_local_attention(const float* __restrict__ qkv,
                                    unsigned short* __restrict__ ctxb,
                                    const int* __restrict__ wptr,
                                    int L, int D, int H) {
    const int idx = blockIdx.x * blockDim.x + threadIdx.x;
    if (idx >= H * L) return;
    const int h = idx / L;
    const int i = idx % L;
    const int W = *wptr;
    const int D3 = 3 * D;

    const float* q = qkv + (size_t)i * D3 + h * LAB_HD;
    float qr[LAB_HD];
#pragma unroll
    for (int f = 0; f < LAB_HD; ++f) qr[f] = q[f];

    int jlo = i - W; if (jlo < 0) jlo = 0;
    int jhi = i + W; if (jhi > L - 1) jhi = L - 1;
    const float scale = 0.125f;  // 1/sqrt(64)

    float m = -3.4028235e38f;
    for (int j = jlo; j <= jhi; ++j) {
        const float* kp = qkv + (size_t)j * D3 + D + h * LAB_HD;
        float s = 0.f;
#pragma unroll
        for (int f = 0; f < LAB_HD; ++f) s += qr[f] * kp[f];
        m = fmaxf(m, s * scale);
    }

    float sum = 0.f;
    float accv[LAB_HD];
#pragma unroll
    for (int f = 0; f < LAB_HD; ++f) accv[f] = 0.f;
    for (int j = jlo; j <= jhi; ++j) {
        const float* kp = qkv + (size_t)j * D3 + D + h * LAB_HD;
        const float* vp = kp + D;
        float s = 0.f;
#pragma unroll
        for (int f = 0; f < LAB_HD; ++f) s += qr[f] * kp[f];
        const float p = __expf(s * scale - m);
        sum += p;
#pragma unroll
        for (int f = 0; f < LAB_HD; ++f) accv[f] += p * vp[f];
    }

    const float inv = 1.f / sum;
    unsigned short* outp = ctxb + (size_t)i * D + h * LAB_HD;
#pragma unroll
    for (int f = 0; f < LAB_HD; ++f) outp[f] = f2bf(accv[f] * inv);
}

// ---------------------------------------------------------------------------
// Fused residual-add + LayerNorm.  One wave (32 threads) per row.
// ---------------------------------------------------------------------------
__global__ void lab_add_layernorm(const float* __restrict__ a,
                                  const float* __restrict__ b,
                                  const float* __restrict__ w,
                                  const float* __restrict__ bias,
                                  float* __restrict__ outF,
                                  unsigned short* __restrict__ outB, int D) {
    const int row  = blockIdx.x;
    const int lane = threadIdx.x;  // 0..31, wave32
    const float* pa = a + (size_t)row * D;
    const float* pb = b + (size_t)row * D;

    float s = 0.f;
    for (int c = lane; c < D; c += 32) s += pa[c] + pb[c];
#pragma unroll
    for (int o = 16; o > 0; o >>= 1) s += __shfl_xor(s, o, 32);
    const float mu = s / (float)D;

    float var = 0.f;
    for (int c = lane; c < D; c += 32) {
        float t = pa[c] + pb[c] - mu;
        var += t * t;
    }
#pragma unroll
    for (int o = 16; o > 0; o >>= 1) var += __shfl_xor(var, o, 32);
    const float inv = rsqrtf(var / (float)D + 1e-5f);

    for (int c = lane; c < D; c += 32) {
        float y = (pa[c] + pb[c] - mu) * inv * w[c] + bias[c];
        if (outF) outF[(size_t)row * D + c] = y;
        if (outB) outB[(size_t)row * D + c] = f2bf(y);
    }
}

// ---------------------------------------------------------------------------
// Host orchestration
// ---------------------------------------------------------------------------
extern "C" void kernel_launch(void* const* d_in, const int* in_sizes, int n_in,
                              void* d_out, int out_size, void* d_ws, size_t ws_size,
                              hipStream_t stream) {
    const float* x         = (const float*)d_in[0];
    const float* in_proj_w = (const float*)d_in[1];
    const float* in_proj_b = (const float*)d_in[2];
    const float* out_w     = (const float*)d_in[3];
    const float* out_b     = (const float*)d_in[4];
    const float* ln1_w     = (const float*)d_in[5];
    const float* ln1_b     = (const float*)d_in[6];
    const float* ln2_w     = (const float*)d_in[7];
    const float* ln2_b     = (const float*)d_in[8];
    const float* ff_w1     = (const float*)d_in[9];
    const float* ff_b1     = (const float*)d_in[10];
    const float* ff_w2     = (const float*)d_in[11];
    const float* ff_b2     = (const float*)d_in[12];
    const int*   window    = (const int*)d_in[13];

    const int D  = in_sizes[4];           // 768
    const int L  = in_sizes[0] / D;       // 4096
    const int D3 = in_sizes[2];           // 2304
    const int FF = in_sizes[10];          // 3072
    const int H  = 12;

    char* base = (char*)d_ws;
    size_t off = 0;
    auto carve = [&](size_t bytes) -> char* {
        char* r = base + off;
        off += (bytes + 255) & ~(size_t)255;
        return r;
    };
    unsigned short* xb    = (unsigned short*)carve((size_t)L * D * 2);
    unsigned short* wqkvb = (unsigned short*)carve((size_t)D3 * D * 2);
    unsigned short* woutb = (unsigned short*)carve((size_t)D * D * 2);
    unsigned short* w1b   = (unsigned short*)carve((size_t)FF * D * 2);
    unsigned short* w2b   = (unsigned short*)carve((size_t)D * FF * 2);
    float*          qkv   = (float*)carve((size_t)L * D3 * 4);
    unsigned short* ctxb  = (unsigned short*)carve((size_t)L * D * 2);
    float*          attn  = (float*)carve((size_t)L * D * 4);
    float*          hF    = (float*)carve((size_t)L * D * 4);
    unsigned short* hB    = (unsigned short*)carve((size_t)L * D * 2);
    unsigned short* gB    = (unsigned short*)carve((size_t)L * FF * 2);
    float*          ff2   = (float*)carve((size_t)L * D * 4);

    auto castN = [&](const float* in, unsigned short* out, size_t n) {
        lab_cast_bf16<<<(unsigned)((n + 255) / 256), 256, 0, stream>>>(in, out, (int)n);
    };

    // 1) bf16 copies of x and weights
    castN(x,         xb,    (size_t)L * D);
    castN(in_proj_w, wqkvb, (size_t)D3 * D);
    castN(out_w,     woutb, (size_t)D * D);
    castN(ff_w1,     w1b,   (size_t)FF * D);
    castN(ff_w2,     w2b,   (size_t)D * FF);

    // 2) qkv = x @ in_proj_w^T + in_proj_b        [L, 3D] fp32
    lab_gemm_wmma_tdm<<<dim3(L / 128, D3 / 128), 256, 0, stream>>>(
        xb, wqkvb, in_proj_b, qkv, nullptr, L, D3, D, 0);

    // 3) banded attention -> ctx (bf16)           [L, D]
    {
        int total = H * L;
        lab_local_attention<<<(total + 63) / 64, 64, 0, stream>>>(
            qkv, ctxb, window, L, D, H);
    }

    // 4) attn = ctx @ out_w^T + out_b             [L, D] fp32
    lab_gemm_wmma_tdm<<<dim3(L / 128, D / 128), 256, 0, stream>>>(
        ctxb, woutb, out_b, attn, nullptr, L, D, D, 0);

    // 5) h = LN(x + attn)                         fp32 + bf16
    lab_add_layernorm<<<L, 32, 0, stream>>>(x, attn, ln1_w, ln1_b, hF, hB, D);

    // 6) g = gelu(h @ ff_w1^T + ff_b1)            [L, FF] bf16
    lab_gemm_wmma_tdm<<<dim3(L / 128, FF / 128), 256, 0, stream>>>(
        hB, w1b, ff_b1, nullptr, gB, L, FF, D, 1);

    // 7) ff2 = g @ ff_w2^T + ff_b2                [L, D] fp32
    lab_gemm_wmma_tdm<<<dim3(L / 128, D / 128), 256, 0, stream>>>(
        gB, w2b, ff_b2, ff2, nullptr, L, D, FF, 0);

    // 8) out = LN(h + ff2)                        fp32 -> d_out
    lab_add_layernorm<<<L, 32, 0, stream>>>(hF, ff2, ln2_w, ln2_b,
                                            (float*)d_out, nullptr, D);
}
